// AttentionBlock_2327872275032
// MI455X (gfx1250) — matrix operand order repaired and checked by reading, not compiled, not run on hardware
//
#include <hip/hip_runtime.h>

// ---------------- problem constants ----------------
constexpr int kN   = 2;
constexpr int kH   = 16;
constexpr int kD   = 64;     // head dim
constexpr int kSeq = 2048;
constexpr int kE   = 1024;   // embed dim

// softmax scale: reference uses 1/sqrt(EMBED_DIM) = 1/32; fold log2(e) in for exp2
#define KSCALE 0.04508422f   // (1/32) * log2(e)

typedef __attribute__((ext_vector_type(16))) __bf16 v16bf;
typedef __attribute__((ext_vector_type(8)))  float  v8f;

union BF16x16 {
  v16bf          v;
  unsigned short s[16];
  unsigned int   u[8];
};

// native f32 -> bf16 (lowers to v_cvt_pk_bf16_f32 when paired)
static __device__ __forceinline__ unsigned short f2bfu(float f) {
  union { __bf16 h; unsigned short s; } u;
  u.h = (__bf16)f;
  return u.s;
}

static __device__ __forceinline__ v8f wmma_bf16(v16bf a, v16bf b, v8f c) {
  return __builtin_amdgcn_wmma_f32_16x16x32_bf16(false, a, false, b, (short)0, c,
                                                 false, false);
}

// A fragment (16x32 bf16): lane holds row (lane&15); elems 0..15 = k: kc+16*half+j.
// Two contiguous 16B loads.
static __device__ __forceinline__ v16bf a_frag_u16(const unsigned short* rowp,
                                                   int kc, int half) {
  BF16x16 f;
  const uint4 lo = *reinterpret_cast<const uint4*>(rowp + kc + half * 16);
  const uint4 hi = *reinterpret_cast<const uint4*>(rowp + kc + half * 16 + 8);
  f.u[0] = lo.x; f.u[1] = lo.y; f.u[2] = lo.z; f.u[3] = lo.w;
  f.u[4] = hi.x; f.u[5] = hi.y; f.u[6] = hi.z; f.u[7] = hi.w;
  return f.v;
}

// B fragment (32x16 bf16): lane holds column (lane&15); 16 contiguous k at kc+16*half.
static __device__ __forceinline__ v16bf b_frag_u16(const unsigned short* rowp,
                                                   int kc, int half) {
  BF16x16 f;
  const unsigned short* p = rowp + kc + half * 16;
  const uint4 lo = *reinterpret_cast<const uint4*>(p);
  const uint4 hi = *reinterpret_cast<const uint4*>(p + 8);
  f.u[0] = lo.x; f.u[1] = lo.y; f.u[2] = lo.z; f.u[3] = lo.w;
  f.u[4] = hi.x; f.u[5] = hi.y; f.u[6] = hi.z; f.u[7] = hi.w;
  return f.v;
}

static __device__ __forceinline__ v16bf a_frag_f32(const float* rowp, int kc, int half) {
  v16bf f;
  const float* p = rowp + kc + half * 16;
#pragma unroll
  for (int j = 0; j < 8; ++j) { f[j] = (__bf16)p[j]; f[8 + j] = (__bf16)p[8 + j]; }
  return f;
}

static __device__ __forceinline__ v16bf b_frag_f32(const float* rowp, int kc, int half) {
  v16bf f;
  const float* p = rowp + kc + half * 16;
#pragma unroll
  for (int j = 0; j < 16; ++j) f[j] = (__bf16)p[j];
  return f;
}

// wave-uniform scalar broadcast (moves index math to SALU, keeps EXEC paths clean)
static __device__ __forceinline__ int wave_id() {
  return __builtin_amdgcn_readfirstlane(
      (int)((blockIdx.x * blockDim.x + threadIdx.x) >> 5));
}

// ============================================================================
// Kernel 0: convert Wo (1024x1024 f32) to bf16 once.
// ============================================================================
__global__ void __launch_bounds__(256) cvt_kernel(const float* __restrict__ src,
                                                  unsigned short* __restrict__ dst) {
  const int i = (int)(blockIdx.x * blockDim.x + threadIdx.x) * 8;
  const float4 a = *reinterpret_cast<const float4*>(src + i);
  const float4 b = *reinterpret_cast<const float4*>(src + i + 4);
  union { unsigned short s[8]; uint4 u; } o;
  o.s[0] = f2bfu(a.x); o.s[1] = f2bfu(a.y); o.s[2] = f2bfu(a.z); o.s[3] = f2bfu(a.w);
  o.s[4] = f2bfu(b.x); o.s[5] = f2bfu(b.y); o.s[6] = f2bfu(b.z); o.s[7] = f2bfu(b.w);
  *reinterpret_cast<uint4*>(dst + i) = o.u;
}

// ============================================================================
// Kernel 1: per-head projections Qp/Kp = x @ W^T (bf16, [N,H,SEQ,64]);
//           Vp stored transposed: Vt [N,H,64,SEQ] for the P*V stage.
// One wave = one (which, n, h, 16-token tile). 8 WMMAs per wave.
// ============================================================================
__global__ void __launch_bounds__(256) proj_kernel(
    const float* __restrict__ vals, const float* __restrict__ keysrc,
    const float* __restrict__ qry,
    const float* __restrict__ Wv, const float* __restrict__ Wk,
    const float* __restrict__ Wq,
    unsigned short* __restrict__ Qp, unsigned short* __restrict__ Kp,
    unsigned short* __restrict__ Vt) {
  const int lane = threadIdx.x & 31;
  const int c    = lane & 15;
  const int half = lane >> 4;
  const int gw   = wave_id();

  const int TPM   = kN * kH * (kSeq / 16);  // tiles per matrix
  const int which = gw / TPM;               // 0:q 1:k 2:v (scalar)
  int rem = gw - which * TPM;
  const int n = rem / (kH * (kSeq / 16));
  rem -= n * (kH * (kSeq / 16));
  const int h     = rem / (kSeq / 16);
  const int tbase = (rem - h * (kSeq / 16)) * 16;

  const float* X = (which == 0) ? qry : (which == 1) ? keysrc : vals;
  const float* W = (which == 0) ? Wq  : (which == 1) ? Wk     : Wv;

  // A: 16 token rows, head slice [h*64 .. h*64+63] of embed dim
  const float* xrow = X + (size_t)(n * kSeq + tbase + c) * kE + h * kD;
  const v16bf a0 = a_frag_f32(xrow, 0, half);
  const v16bf a1 = a_frag_f32(xrow, 32, half);

#pragma unroll
  for (int et = 0; et < 4; ++et) {
    // B[d, e] = W[e][d]  -> column e is row e of W (contiguous in d)
    const float* wrow = W + (size_t)(et * 16 + c) * kD;
    v8f acc = {};
    acc = wmma_bf16(a0, b_frag_f32(wrow, 0, half), acc);
    acc = wmma_bf16(a1, b_frag_f32(wrow, 32, half), acc);

    if (which <= 1) {  // Qp / Kp : [N,H,SEQ,64]
      unsigned short* O = ((which == 0) ? Qp : Kp) +
                          (size_t)(n * kH + h) * kSeq * kD;
#pragma unroll
      for (int r = 0; r < 8; ++r) {
        const int token = tbase + r + 8 * half;               // D row M
        O[(size_t)token * kD + et * 16 + c] = f2bfu(acc[r]);  // D col N = e
      }
    } else {           // Vt : [N,H,64,SEQ]
      unsigned short* O =
          Vt + (size_t)((n * kH + h) * kD + et * 16 + c) * kSeq;
#pragma unroll
      for (int r = 0; r < 8; ++r) O[tbase + r + 8 * half] = f2bfu(acc[r]);
    }
  }
}

// ============================================================================
// Kernel 2: flash attention, S^T formulation.
// One wave = (n, h, 16-q tile). Streams keys in 32-wide chunks:
//   S^T(16k x 16q) = K_rows x Q^T   (2 WMMAs per 16-k tile)
//   O^T(64d x 16q) += Vt x P^T      (4 WMMAs per 32-k chunk)
// Softmax axis (k) lives in each lane's 8 slots + one shfl_xor(16).
// ============================================================================
__global__ void __launch_bounds__(256) attn_kernel(
    const unsigned short* __restrict__ Qp, const unsigned short* __restrict__ Kp,
    const unsigned short* __restrict__ Vt, const int* __restrict__ mask,
    unsigned short* __restrict__ AO) {
  const int lane = threadIdx.x & 31;
  const int c    = lane & 15;
  const int half = lane >> 4;
  const int gw   = wave_id();

  const int n = gw / (kH * (kSeq / 16));
  int rem = gw - n * (kH * (kSeq / 16));
  const int h     = rem / (kSeq / 16);
  const int qbase = (rem - h * (kSeq / 16)) * 16;

  const size_t hoff = (size_t)(n * kH + h) * kSeq * kD;
  const unsigned short* Qh = Qp + hoff;
  const unsigned short* Kh = Kp + hoff;
  const unsigned short* Vh = Vt + hoff;  // [64][SEQ]

  // Q as B operand: column = this lane's q row, contiguous d
  const unsigned short* qrow = Qh + (size_t)(qbase + c) * kD;
  const v16bf bq0 = b_frag_u16(qrow, 0, half);
  const v16bf bq1 = b_frag_u16(qrow, 32, half);

  // mask row for this lane's q (k is contiguous)
  const int* mrow = mask + (size_t)(n * kSeq + qbase + c) * kSeq;

  float m_i = -3.0e38f, l_i = 0.0f;
  v8f o[4] = {v8f{}, v8f{}, v8f{}, v8f{}};

  for (int kt = 0; kt < kSeq / 32; ++kt) {
    const int kb = kt * 32;

    // prefetch next chunk's K rows + mask row while this chunk computes
    if (kt + 1 < kSeq / 32) {
      __builtin_prefetch(Kh + (size_t)(kb + 32 + c) * kD, 0, 0);
      __builtin_prefetch(Kh + (size_t)(kb + 48 + c) * kD, 0, 0);
      __builtin_prefetch(mrow + kb + 32, 0, 0);
    }

    float s0[8], s1[8];
#pragma unroll
    for (int tt = 0; tt < 2; ++tt) {
      const int kbb = kb + tt * 16;
      const unsigned short* krow = Kh + (size_t)(kbb + c) * kD;  // A: key rows
      v8f acc = {};
      acc = wmma_bf16(a_frag_u16(krow, 0, half), bq0, acc);
      acc = wmma_bf16(a_frag_u16(krow, 32, half), bq1, acc);
      const int* mp = mrow + kbb + 8 * half;  // k = kbb + 8*half + r
      float* s = tt ? s1 : s0;
#pragma unroll
      for (int r = 0; r < 8; ++r) s[r] = (mp[r] == 0) ? -1.0e20f : acc[r];
    }

    // ---- online softmax update (per q column; halves exchange via xor 16) ----
    float mloc = s0[0];
#pragma unroll
    for (int r = 1; r < 8; ++r) mloc = fmaxf(mloc, s0[r]);
#pragma unroll
    for (int r = 0; r < 8; ++r) mloc = fmaxf(mloc, s1[r]);
    mloc = fmaxf(mloc, __shfl_xor(mloc, 16, 32));
    const float mnew = fmaxf(m_i, mloc);
    const float corr = exp2f((m_i - mnew) * KSCALE);
    m_i = mnew;
    l_i *= corr;
#pragma unroll
    for (int dt = 0; dt < 4; ++dt)
#pragma unroll
      for (int r = 0; r < 8; ++r) o[dt][r] *= corr;

    unsigned short p0u[8], p1u[8];
    float psum = 0.0f;
#pragma unroll
    for (int r = 0; r < 8; ++r) {
      const float p0 = exp2f((s0[r] - mnew) * KSCALE);
      const float p1 = exp2f((s1[r] - mnew) * KSCALE);
      psum += p0 + p1;
      p0u[r] = f2bfu(p0);
      p1u[r] = f2bfu(p1);
    }
    psum += __shfl_xor(psum, 16, 32);
    l_i += psum;

    // ---- redistribute D-fragment -> B-fragment: single lane c <-> c^16 swap ----
    // lane half0 needs tile0 rows 8..15 (partner's p0); half1 needs tile1 rows 0..7
    // (partner's p1). Each lane sends (half ? p0 : p1).
    unsigned int sp[4], rp[4];
#pragma unroll
    for (int w = 0; w < 4; ++w) {
      const unsigned short a = half ? p0u[2 * w]     : p1u[2 * w];
      const unsigned short b = half ? p0u[2 * w + 1] : p1u[2 * w + 1];
      sp[w] = (unsigned int)a | ((unsigned int)b << 16);
      rp[w] = (unsigned int)__shfl_xor((int)sp[w], 16, 32);
    }
    BF16x16 pb;
#pragma unroll
    for (int w = 0; w < 4; ++w) {
      const unsigned short ra = (unsigned short)(rp[w] & 0xFFFFu);
      const unsigned short rb = (unsigned short)(rp[w] >> 16);
      // elems 0..7  = k_local (16*half + 0..7);  elems 8..15 = k_local (16*half + 8..15)
      pb.s[2 * w]     = half ? ra : p0u[2 * w];
      pb.s[2 * w + 1] = half ? rb : p0u[2 * w + 1];
      pb.s[8 + 2 * w]     = half ? p1u[2 * w]     : ra;
      pb.s[8 + 2 * w + 1] = half ? p1u[2 * w + 1] : rb;
    }

    // ---- O^T += Vt(A) x P^T(B) over this 32-k chunk ----
#pragma unroll
    for (int dt = 0; dt < 4; ++dt) {
      const unsigned short* vrow = Vh + (size_t)(dt * 16 + c) * kSeq;
      o[dt] = wmma_bf16(a_frag_u16(vrow, kb, half), pb.v, o[dt]);
    }
  }

  // ---- normalize and store O[q, h*64 + d] as bf16 into AO [N*SEQ, 1024] ----
  const float inv = 1.0f / l_i;
  unsigned short* orow = AO + (size_t)(n * kSeq + qbase + c) * kE + h * kD;
#pragma unroll
  for (int dt = 0; dt < 4; ++dt)
#pragma unroll
    for (int r = 0; r < 8; ++r)
      orow[dt * 16 + 8 * half + r] = f2bfu(o[dt][r] * inv);
}

// ============================================================================
// Kernel 3: out = AO @ Wo^T + bo  (fp32 result). One wave = one 16x16 tile,
// K = 1024 -> 32 WMMAs.  Wo pre-converted to bf16 (Wob).
// ============================================================================
__global__ void __launch_bounds__(256) out_kernel(
    const unsigned short* __restrict__ AO, const unsigned short* __restrict__ Wob,
    const float* __restrict__ bo, float* __restrict__ out) {
  const int lane = threadIdx.x & 31;
  const int c    = lane & 15;
  const int half = lane >> 4;
  const int gw   = wave_id();

  const int tt = gw >> 6;          // token tile 0..255
  const int et = gw & 63;          // e tile 0..63

  const unsigned short* arow = AO  + (size_t)(tt * 16 + c) * kE;  // A: token rows
  const unsigned short* wrow = Wob + (size_t)(et * 16 + c) * kE;  // B col e = Wo row e

  v8f acc = {};
  for (int kc = 0; kc < kE; kc += 32)
    acc = wmma_bf16(a_frag_u16(arow, kc, half), b_frag_u16(wrow, kc, half), acc);

  const float bias = bo[et * 16 + c];
  float* obase = out + (size_t)(tt * 16) * kE + et * 16 + c;
#pragma unroll
  for (int r = 0; r < 8; ++r)
    obase[(size_t)(r + 8 * half) * kE] = acc[r] + bias;
}

// ============================================================================
extern "C" void kernel_launch(void* const* d_in, const int* in_sizes, int n_in,
                              void* d_out, int out_size, void* d_ws, size_t ws_size,
                              hipStream_t stream) {
  (void)in_sizes; (void)n_in; (void)out_size; (void)ws_size;
  const float* values = (const float*)d_in[0];
  const float* keys   = (const float*)d_in[1];
  const float* query  = (const float*)d_in[2];
  const int*   mask   = (const int*)d_in[3];
  const float* Wv     = (const float*)d_in[4];
  const float* Wk     = (const float*)d_in[5];
  const float* Wq     = (const float*)d_in[6];
  const float* Wo     = (const float*)d_in[7];
  const float* bo     = (const float*)d_in[8];
  float* out = (float*)d_out;

  // workspace: Qp | Kp | Vt (each 8 MB) | AO (8 MB) | Wob (2 MB)  = 34 MB bf16
  const size_t headElems = (size_t)kN * kH * kSeq * kD;
  unsigned short* Qp  = (unsigned short*)d_ws;
  unsigned short* Kp  = Qp + headElems;
  unsigned short* Vt  = Kp + headElems;
  unsigned short* AO  = Vt + headElems;                 // kN*kSeq*kE
  unsigned short* Wob = AO + (size_t)kN * kSeq * kE;    // kE*kE

  // Wo f32 -> bf16: 1M elems / (256 thr * 8) = 512 blocks
  cvt_kernel<<<512, 256, 0, stream>>>(Wo, Wob);
  // proj: 3 * 2*16*128 = 12288 waves -> 1536 blocks of 8 waves
  proj_kernel<<<1536, 256, 0, stream>>>(values, keys, query, Wv, Wk, Wq, Qp, Kp, Vt);
  // attn: 2*16*128 = 4096 waves -> 512 blocks
  attn_kernel<<<512, 256, 0, stream>>>(Qp, Kp, Vt, mask, AO);
  // out: 256 token tiles * 64 e tiles = 16384 waves -> 2048 blocks
  out_kernel<<<2048, 256, 0, stream>>>(AO, Wob, bo, out);
}
